// TripletRatio_60129542144775
// MI455X (gfx1250) — matrix operand order repaired and checked by reading, not compiled, not run on hardware
//
#include <hip/hip_runtime.h>

typedef __attribute__((ext_vector_type(16))) __bf16 v16bf;
typedef __attribute__((ext_vector_type(8)))  float  v8f;
typedef __attribute__((ext_vector_type(4)))  unsigned u32x4;
typedef __attribute__((ext_vector_type(4)))  int      v4i;

#define GN 8192
#define GK 256
#define NCLS 100
#define TM 64            // rows per block
#define TN 64            // cols per iteration (two 16-wide subtiles per wave)
#define LSTR 264         // LDS row stride in halfs (528B) -> conflict-free b128 reads
#define NBLK (GN / TM)   // 128 blocks
#define NT (GN / TN)     // 128 column iterations

__device__ __forceinline__ unsigned short f32_to_bf16_rne(float f) {
    unsigned u = __builtin_bit_cast(unsigned, f);
    unsigned r = u + 0x7FFFu + ((u >> 16) & 1u);
    return (unsigned short)(r >> 16);
}
__device__ __forceinline__ float bf16_to_f32(unsigned short h) {
    unsigned u = ((unsigned)h) << 16;
    return __builtin_bit_cast(float, u);
}

union Frag { u32x4 q[2]; v16bf v; };

__device__ __forceinline__ void wait_async0() {
#if __has_builtin(__builtin_amdgcn_s_wait_asynccnt)
    __builtin_amdgcn_s_wait_asynccnt(0);
#else
    asm volatile("s_wait_asynccnt 0x0" ::: "memory");
#endif
}

// copy one 16B chunk global(bf16 plane) -> LDS, async if available
__device__ __forceinline__ void cp16(const unsigned short* g, unsigned short* l) {
#if __has_builtin(__builtin_amdgcn_global_load_async_to_lds_b128)
    __builtin_amdgcn_global_load_async_to_lds_b128(
        (__attribute__((address_space(1))) v4i*)(g),
        (__attribute__((address_space(3))) v4i*)(l), 0, 0);
#else
    *(u32x4*)l = *(const u32x4*)g;
#endif
}

// fill one B buffer: 64 rows x 256 halfs, hi+lo planes; 256 threads, 16 chunks each
__device__ __forceinline__ void fill_b(unsigned short* dHi, unsigned short* dLo,
                                       const unsigned short* __restrict__ Ehi,
                                       const unsigned short* __restrict__ Elo,
                                       int cBase, int t) {
    const int row = t >> 2, q = t & 3;              // 4 threads per row, 128B each
    const size_t gb = (size_t)(cBase + row) * GK + q * 64;
    const int lb = row * LSTR + q * 64;
#pragma unroll
    for (int k = 0; k < 8; ++k) {
        cp16(Ehi + gb + k * 8, dHi + lb + k * 8);
        cp16(Elo + gb + k * 8, dLo + lb + k * 8);
    }
}

// ---------------- one-pass f32 -> bf16 hi/lo planes ------------------------------
__global__ __launch_bounds__(256) void prep_kernel(const float* __restrict__ emb,
                                                   unsigned short* __restrict__ Ehi,
                                                   unsigned short* __restrict__ Elo) {
    int i = (blockIdx.x * 256 + threadIdx.x) * 2;
    float x0 = emb[i], x1 = emb[i + 1];
    unsigned short h0 = f32_to_bf16_rne(x0), h1 = f32_to_bf16_rne(x1);
    unsigned short l0 = f32_to_bf16_rne(x0 - bf16_to_f32(h0));
    unsigned short l1 = f32_to_bf16_rne(x1 - bf16_to_f32(h1));
    ((unsigned*)Ehi)[i >> 1] = (unsigned)h0 | ((unsigned)h1 << 16);
    ((unsigned*)Elo)[i >> 1] = (unsigned)l0 | ((unsigned)l1 << 16);
}

// ---------------- exact fp32 row squared norms -----------------------------------
__global__ __launch_bounds__(256) void rowsq_kernel(const float* __restrict__ emb,
                                                    float* __restrict__ sq) {
    int row  = blockIdx.x * 8 + (threadIdx.x >> 5);
    int lane = threadIdx.x & 31;
    const float* p = emb + (size_t)row * GK;
    float s = 0.f;
#pragma unroll
    for (int i = 0; i < 8; ++i) { float x = p[lane + 32 * i]; s += x * x; }
#pragma unroll
    for (int m = 16; m; m >>= 1) s += __shfl_xor(s, m, 32);
    if (lane == 0) sq[row] = s;
}

// ---------------- fused Gram + loss, double-buffered async B ---------------------
__global__ __launch_bounds__(256) void gram_loss_kernel(
    const unsigned short* __restrict__ Ehi, const unsigned short* __restrict__ Elo,
    const int* __restrict__ labels, const float* __restrict__ sq,
    float* __restrict__ partials) {
    __shared__ unsigned short sAhi[TM * LSTR];
    __shared__ unsigned short sAlo[TM * LSTR];
    __shared__ unsigned short sBhi[2][TN * LSTR];
    __shared__ unsigned short sBlo[2][TN * LSTR];
    __shared__ float sRed[2][2][TM];

    const int tid  = threadIdx.x;
    const int lane = tid & 31;
    const int wave = tid >> 5;
    const int wr   = wave >> 1;   // 0..3
    const int wc   = wave & 1;    // 0..1 (32-col half; 2 subtiles)
    const int hi16 = lane >> 4;
    const int cl   = lane & 15;
    const int rowT = blockIdx.x * TM;

    // A tile (64 x 256 hi/lo) -> LDS (plain copy, once)
    {
        const int row = tid >> 2, q = tid & 3;
        const size_t gb = (size_t)(rowT + row) * GK + q * 64;
        const int lb = row * LSTR + q * 64;
#pragma unroll
        for (int k = 0; k < 8; ++k) {
            *(u32x4*)&sAhi[lb + k * 8] = *(const u32x4*)&Ehi[gb + k * 8];
            *(u32x4*)&sAlo[lb + k * 8] = *(const u32x4*)&Elo[gb + k * 8];
        }
    }

    float sqR[8]; int lblR[8];
#pragma unroll
    for (int j = 0; j < 8; ++j) {
        int r = rowT + wr * 16 + hi16 * 8 + j;
        sqR[j] = sq[r];
        lblR[j] = labels[r];
    }
    float numA[8], denA[8];
#pragma unroll
    for (int j = 0; j < 8; ++j) { numA[j] = 0.f; denA[j] = 0.f; }

    fill_b(sBhi[0], sBlo[0], Ehi, Elo, 0, tid);   // prefill buffer 0
    wait_async0();
    __syncthreads();

    for (int ct = 0; ct < NT; ++ct) {
        const int cur = ct & 1;
        if (ct + 1 < NT)                           // prefetch next tile (hidden)
            fill_b(sBhi[cur ^ 1], sBlo[cur ^ 1], Ehi, Elo, (ct + 1) * TN, tid);

        const unsigned short* bH = sBhi[cur];
        const unsigned short* bL = sBlo[cur];
        v8f acc0 = {}, acc1 = {};
#pragma unroll
        for (int ks = 0; ks < GK / 32; ++ks) {
            Frag ahi, alo, b0h, b0l, b1h, b1l;
            const unsigned short* pa = &sAhi[(wr * 16 + cl) * LSTR + ks * 32 + hi16 * 16];
            const unsigned short* qa = &sAlo[(wr * 16 + cl) * LSTR + ks * 32 + hi16 * 16];
            ahi.q[0] = *(const u32x4*)pa;  ahi.q[1] = *(const u32x4*)(pa + 8);
            alo.q[0] = *(const u32x4*)qa;  alo.q[1] = *(const u32x4*)(qa + 8);
            const unsigned short* pb = &bH[(wc * 32 + cl) * LSTR + ks * 32 + hi16 * 16];
            const unsigned short* qb = &bL[(wc * 32 + cl) * LSTR + ks * 32 + hi16 * 16];
            b0h.q[0] = *(const u32x4*)pb;               b0h.q[1] = *(const u32x4*)(pb + 8);
            b0l.q[0] = *(const u32x4*)qb;               b0l.q[1] = *(const u32x4*)(qb + 8);
            b1h.q[0] = *(const u32x4*)(pb + 16 * LSTR); b1h.q[1] = *(const u32x4*)(pb + 16 * LSTR + 8);
            b1l.q[0] = *(const u32x4*)(qb + 16 * LSTR); b1l.q[1] = *(const u32x4*)(qb + 16 * LSTR + 8);

            // two independent 3-term split chains, interleaved for pipelining
            acc0 = __builtin_amdgcn_wmma_f32_16x16x32_bf16(false, ahi.v, false, b0h.v, (short)0, acc0, false, false);
            acc1 = __builtin_amdgcn_wmma_f32_16x16x32_bf16(false, ahi.v, false, b1h.v, (short)0, acc1, false, false);
            acc0 = __builtin_amdgcn_wmma_f32_16x16x32_bf16(false, ahi.v, false, b0l.v, (short)0, acc0, false, false);
            acc1 = __builtin_amdgcn_wmma_f32_16x16x32_bf16(false, ahi.v, false, b1l.v, (short)0, acc1, false, false);
            acc0 = __builtin_amdgcn_wmma_f32_16x16x32_bf16(false, alo.v, false, b0h.v, (short)0, acc0, false, false);
            acc1 = __builtin_amdgcn_wmma_f32_16x16x32_bf16(false, alo.v, false, b1h.v, (short)0, acc1, false, false);
        }

        // fused epilogue for both subtiles
        const int c0 = ct * TN + wc * 32 + cl;
        const int c1 = c0 + 16;
        const float sqC0 = sq[c0], sqC1 = sq[c1];
        const int lC0 = labels[c0], lC1 = labels[c1];
#pragma unroll
        for (int j = 0; j < 8; ++j) {
            float r0 = sqR[j] + sqC0 - 2.0f * acc0[j];
            float d0 = __builtin_amdgcn_sqrtf(fmaxf(r0, 1e-4f));
            if (lblR[j] == lC0) numA[j] += d0;
            else                denA[j] += __builtin_amdgcn_rcpf(d0 + 0.1f);
            float r1 = sqR[j] + sqC1 - 2.0f * acc1[j];
            float d1 = __builtin_amdgcn_sqrtf(fmaxf(r1, 1e-4f));
            if (lblR[j] == lC1) numA[j] += d1;
            else                denA[j] += __builtin_amdgcn_rcpf(d1 + 0.1f);
        }

        wait_async0();        // next buffer landed (hidden behind the 48 WMMAs)
        __syncthreads();      // everyone done reading cur before it is refilled
    }

    // lane-group reduction (rows j / j+8), then combine the two wc halves
#pragma unroll
    for (int j = 0; j < 8; ++j) {
        float n = numA[j], dn = denA[j];
#pragma unroll
        for (int m = 8; m; m >>= 1) {
            n += __shfl_xor(n, m, 32);
            dn += __shfl_xor(dn, m, 32);
        }
        if (cl == 0) {
            int r = wr * 16 + hi16 * 8 + j;
            sRed[0][wc][r] = n;
            sRed[1][wc][r] = dn;
        }
    }
    __syncthreads();
    if (tid < TM) {
        float n  = sRed[0][0][tid] + sRed[0][1][tid];
        float dn = sRed[1][0][tid] + sRed[1][1][tid];
        sRed[0][0][tid] = n * dn;       // per-row loss contribution
    }
    __syncthreads();
    if (tid == 0) {
        float L = 0.f;
        for (int i = 0; i < TM; ++i) L += sRed[0][0][i];
        partials[blockIdx.x] = L;
    }
}

// ---------------- finalize: label histogram m_sum + scalar division --------------
__global__ __launch_bounds__(256) void finalize_kernel(const float* __restrict__ partials,
                                                       const int* __restrict__ labels,
                                                       float* __restrict__ out) {
    __shared__ double sL[256];
    __shared__ int cnt[NCLS];
    int t = threadIdx.x;
    if (t < NCLS) cnt[t] = 0;
    __syncthreads();
    for (int i = t; i < GN; i += 256) atomicAdd(&cnt[labels[i]], 1);
    sL[t] = (t < NBLK) ? (double)partials[t] : 0.0;
    __syncthreads();
    for (int s = 128; s > 0; s >>= 1) {
        if (t < s) sL[t] += sL[t + s];
        __syncthreads();
    }
    if (t == 0) {
        double M = 0.0;   // m_sum = sum_l c_l^2 * (N - c_l)
        for (int c = 0; c < NCLS; ++c) {
            double cc = (double)cnt[c];
            M += cc * cc * ((double)GN - cc);
        }
        out[0] = (float)(sL[0] / M);
    }
}

extern "C" void kernel_launch(void* const* d_in, const int* in_sizes, int n_in,
                              void* d_out, int out_size, void* d_ws, size_t ws_size,
                              hipStream_t stream) {
    (void)in_sizes; (void)n_in; (void)out_size; (void)ws_size;
    const float* emb   = (const float*)d_in[0];
    const int* labels  = (const int*)d_in[1];

    unsigned short* Ehi = (unsigned short*)d_ws;                 // 4 MB
    unsigned short* Elo = Ehi + (size_t)GN * GK;                 // 4 MB
    float* sq           = (float*)(Elo + (size_t)GN * GK);       // 32 KB
    float* partials     = sq + GN;                               // 512 B

    prep_kernel<<<(GN * GK) / (2 * 256), 256, 0, stream>>>(emb, Ehi, Elo);
    rowsq_kernel<<<GN / 8, 256, 0, stream>>>(emb, sq);
    gram_loss_kernel<<<NBLK, 256, 0, stream>>>(Ehi, Elo, labels, sq, partials);
    finalize_kernel<<<1, 256, 0, stream>>>(partials, labels, (float*)d_out);
}
